// SharedEncoder_19628000543338
// MI455X (gfx1250) — matrix operand order repaired and checked by reading, not compiled, not run on hardware
//
#include <hip/hip_runtime.h>
#include <hip/hip_bf16.h>
#include <math.h>

typedef __attribute__((ext_vector_type(16))) _Float16 v16h;
typedef __attribute__((ext_vector_type(8)))  float    v8f;

#define NND    32768
#define NEDGE  524288
#define NBAT   64
#define NPGR   512
#define DDIM   128
#define NHEAD  4
#define DHEAD  32
#define NLAY   4
#define LNEPS  1e-5f
#define NSLOPE 0.01f

#ifndef __has_builtin
#define __has_builtin(x) 0
#endif
#if __has_builtin(__builtin_amdgcn_tensor_load_to_lds) && \
    __has_builtin(__builtin_amdgcn_s_wait_tensorcnt)
#define USE_TDM 1
#else
#define USE_TDM 0
#endif

// ---------------------------------------------------------------- WMMA helpers
__device__ __forceinline__ v8f wmma16(v16h a, v16h b, v8f c) {
  return __builtin_amdgcn_wmma_f32_16x16x32_f16(false, a, false, b, (short)0, c,
                                                false, false);
}

// A-matrix 16x32 f16 fragment (ISA 7.12.2): lane L -> row L%16,
// K = {(L/16)*8 .. +7} and {(L/16)*8+16 .. +23}, contiguous dword pairs.
__device__ __forceinline__ v16h ld_afrag(const _Float16* __restrict__ p,
                                         int stride, int lane) {
  const int m = lane & 15, kb = (lane >> 4) * 8;
  union { v16h v; unsigned u[8]; } r;
  const unsigned* a0 = (const unsigned*)(p + m * stride + kb);
  const unsigned* a1 = (const unsigned*)(p + m * stride + kb + 16);
#pragma unroll
  for (int i = 0; i < 4; ++i) { r.u[i] = a0[i]; r.u[4 + i] = a1[i]; }
  return r.v;
}

// B-matrix 32x16 f16 fragment: lane L -> col nbase + L%16,
// K = {(L/16)*8 .. +7} and {(L/16)*8+16 .. +23}. LDS tile stored [n][k].
__device__ __forceinline__ v16h ld_bfrag(const _Float16* __restrict__ p,
                                         int stride, int lane, int nbase) {
  const int n = nbase + (lane & 15), kb = (lane >> 4) * 8;
  union { v16h v; unsigned u[8]; } r;
  const unsigned* b0 = (const unsigned*)(p + n * stride + kb);
  const unsigned* b1 = (const unsigned*)(p + n * stride + kb + 16);
#pragma unroll
  for (int i = 0; i < 4; ++i) { r.u[i] = b0[i]; r.u[4 + i] = b1[i]; }
  return r.v;
}

// 16-lane butterfly reductions on DPP16 (pure VALU, no LDS):
// xor1 = quad_perm{1,0,3,2}, xor2 = quad_perm{2,3,0,1},
// quad->oct = row_half_mirror (0x141), oct->row16 = row_mirror (0x140).
template <int CTRL>
__device__ __forceinline__ float dpp_bcast(float v) {
  const int x =
      __builtin_amdgcn_update_dpp(0, __float_as_int(v), CTRL, 0xF, 0xF, true);
  return __int_as_float(x);
}
__device__ __forceinline__ float red16_sum(float v) {
  v += dpp_bcast<0xB1>(v);
  v += dpp_bcast<0x4E>(v);
  v += dpp_bcast<0x141>(v);
  v += dpp_bcast<0x140>(v);
  return v;
}
__device__ __forceinline__ float red16_max(float v) {
  v = fmaxf(v, dpp_bcast<0xB1>(v));
  v = fmaxf(v, dpp_bcast<0x4E>(v));
  v = fmaxf(v, dpp_bcast<0x141>(v));
  v = fmaxf(v, dpp_bcast<0x140>(v));
  return v;
}

// ---------------------------------------------------------------- TDM staging
#if USE_TDM
typedef __attribute__((ext_vector_type(4))) unsigned u32x4;
typedef __attribute__((ext_vector_type(8))) int      i32x8;
typedef __attribute__((ext_vector_type(4))) int      i32x4;

// Load a [rows x 32] f16 tile (row stride = strideElems in memory) into LDS at
// byte offset ldsOff, padding 4 DWORDs after every 16 (LDS row stride = 40
// halves, matching ld_afrag/ld_bfrag). D# per CDNA5 ISA sec. 8.
__device__ __forceinline__ void tdm_load_2d(unsigned ldsOff, const void* g,
                                            int rows, int strideElems) {
  const unsigned long long ga = (unsigned long long)(size_t)g;
  u32x4 g0;
  g0[0] = 1u;                                   // count=1, user descriptor
  g0[1] = ldsOff;                               // lds_addr
  g0[2] = (unsigned)ga;                         // global_addr[31:0]
  g0[3] = (unsigned)(ga >> 32) | (2u << 30);    // global_addr[56:32] | type=2
  i32x8 g1;
  g1[0] = (1 << 16)       // data_size = 2B
        | (1 << 20)       // pad_enable
        | (3 << 22)       // pad_interval: every 16 DWORDs
        | (3 << 25);      // pad_amount: 4 DWORDs
  g1[1] = (strideElems & 0xFFFF) << 16;                        // tensor_dim0 lo
  g1[2] = ((strideElems >> 16) & 0xFFFF) | ((rows & 0xFFFF) << 16);
  g1[3] = (32 << 16);                                          // tile_dim0 = 32
  g1[4] = rows & 0xFFFF;                                       // tile_dim1
  g1[5] = strideElems;                                         // dim0_stride lo
  g1[6] = 0;
  g1[7] = 0;
  const i32x4 z4 = {};
  const i32x8 z8 = {};
  __builtin_amdgcn_tensor_load_to_lds(g0, g1, z4, z4, z8, 0);
}
#endif

__device__ __forceinline__ unsigned lds_off(const void* p) {
  return (unsigned)(size_t)p;  // generic LDS ptr: addr[31:0] = LDS offset
}

// ---------------------------------------------------------------- small kernels
__global__ void k_cvt(const float* __restrict__ in, _Float16* __restrict__ o, int n) {
  int i = blockIdx.x * 256 + threadIdx.x;
  if (i < n) o[i] = (_Float16)in[i];
}

__global__ void k_zero4(float4* __restrict__ p, int n4) {
  int i = blockIdx.x * 256 + threadIdx.x;
  if (i < n4) p[i] = make_float4(0.f, 0.f, 0.f, 0.f);
}

// h = x @ node_W.T + node_b   (k = 7; VALU, memory bound)
__global__ __launch_bounds__(256) void k_node(const float* __restrict__ x,
                                              const float* __restrict__ W,
                                              const float* __restrict__ b,
                                              float* __restrict__ h) {
  const int t = threadIdx.x;
  const int n = blockIdx.x * 2 + (t >> 7);
  const int d = t & 127;
  float s = b[d];
#pragma unroll
  for (int j = 0; j < 7; ++j) s += x[n * 7 + j] * W[d * 7 + j];
  h[(size_t)n * DDIM + d] = s;
}

// aggr[dst] += relu(h[src] + e)   one wave per edge, 4 cols/lane, f32 atomics
__global__ __launch_bounds__(256) void k_msg(const float* __restrict__ h,
                                             const _Float16* __restrict__ e,
                                             const int* __restrict__ ei,
                                             float* __restrict__ aggr) {
  const int lane = threadIdx.x & 31, wave = threadIdx.x >> 5;
  const int edge = blockIdx.x * 8 + wave;
  const int src = ei[edge];
  const int dst = ei[NEDGE + edge];
  const int c = lane * 4;
  const float4 hv = *(const float4*)(h + (size_t)src * DDIM + c);
  const _Float16* ep = e + (size_t)edge * DDIM + c;
  float m0 = fmaxf(hv.x + (float)ep[0], 0.f);
  float m1 = fmaxf(hv.y + (float)ep[1], 0.f);
  float m2 = fmaxf(hv.z + (float)ep[2], 0.f);
  float m3 = fmaxf(hv.w + (float)ep[3], 0.f);
  float* ap = aggr + (size_t)dst * DDIM + c;
  atomicAdd(ap + 0, m0);
  atomicAdd(ap + 1, m1);
  atomicAdd(ap + 2, m2);
  atomicAdd(ap + 3, m3);
}

// ---------------------------------------------------------------- generic WMMA GEMM
// C[M,Nout] = epi( act_src(A)[M,K] @ W16[Nout,K]^T + bias )
// ASRC: 0 = f16 A (TDM-staged); 1 = f32 Af0 (+Af1 summed); 2 = edge stage-1
// EPI : 0 store f16; 1 relu->f16; 2 gelu->f16;
//       3 LN+leakyReLU, out32 = resid + y (in-place h update);
//       4 LN(resid + y) -> out32 f32 (+ optional f16 copy)
template <int ASRC, int EPI>
__global__ __launch_bounds__(256) void k_gemm(
    const _Float16* __restrict__ A16, const float* __restrict__ Af0,
    const float* __restrict__ Af1, const float* __restrict__ ea,
    const float* __restrict__ eaW1, const float* __restrict__ eaB1,
    const _Float16* __restrict__ W16, const float* __restrict__ bias, int K,
    int ldOut, _Float16* __restrict__ out16, float* __restrict__ out32,
    const float* __restrict__ resid, const float* __restrict__ lng,
    const float* __restrict__ lnb) {
  __shared__ _Float16 At[128 * 40];
  __shared__ _Float16 Wt[128 * 40];
  const int lane = threadIdx.x & 31, wave = threadIdx.x >> 5;
  const int hb = lane >> 4, ln16 = lane & 15;
  const int rowBlk = blockIdx.x * 128;
  const int colBase = blockIdx.y * 128;
  const int sr = threadIdx.x >> 1;         // staging row 0..127
  const int sk0 = (threadIdx.x & 1) * 16;  // staging k half

  float earow[8];
  if constexpr (ASRC == 2) {
    const float* e = ea + (size_t)(rowBlk + sr) * 8;
#pragma unroll
    for (int j = 0; j < 8; ++j) earow[j] = e[j];
  }

  v8f acc[8] = {};

  for (int k0 = 0; k0 < K; k0 += 32) {
    __syncthreads();
#if USE_TDM
    if (wave == 0) {  // DMA uniform tiles via Tensor Data Mover
      if constexpr (ASRC == 0)
        tdm_load_2d(lds_off(At), A16 + (size_t)rowBlk * K + k0, 128, K);
      tdm_load_2d(lds_off(Wt), W16 + (size_t)colBase * K + k0, 128, K);
    }
#endif
    if constexpr (ASRC == 2) {  // fused edge-encoder stage 1 -> LDS
#pragma unroll
      for (int kk = 0; kk < 16; ++kk) {
        const int k = k0 + sk0 + kk;
        float s = eaB1[k];
#pragma unroll
        for (int j = 0; j < 8; ++j) s += earow[j] * eaW1[k * 8 + j];
        At[sr * 40 + sk0 + kk] = (_Float16)fmaxf(s, 0.0f);
      }
    } else if constexpr (ASRC == 1) {
      const size_t ro = (size_t)(rowBlk + sr) * K + k0 + sk0;
#pragma unroll
      for (int kk = 0; kk < 16; ++kk) {
        const float v = Af1 ? (Af0[ro + kk] + Af1[ro + kk]) : Af0[ro + kk];
        At[sr * 40 + sk0 + kk] = (_Float16)v;
      }
    }
#if !USE_TDM
    if constexpr (ASRC == 0) {
      const size_t ro = (size_t)(rowBlk + sr) * K + k0 + sk0;
#pragma unroll
      for (int kk = 0; kk < 16; ++kk)
        At[sr * 40 + sk0 + kk] = A16[ro + kk];
    }
    {
      const size_t wo = (size_t)(colBase + sr) * K + k0 + sk0;
#pragma unroll
      for (int kk = 0; kk < 16; ++kk) Wt[sr * 40 + sk0 + kk] = W16[wo + kk];
    }
#else
    if (wave == 0) __builtin_amdgcn_s_wait_tensorcnt(0);
#endif
    __syncthreads();
    const v16h af = ld_afrag(At + wave * 16 * 40, 40, lane);
#pragma unroll
    for (int t = 0; t < 8; ++t) {
      const v16h bf = ld_bfrag(Wt, 40, lane, t * 16);
      acc[t] = wmma16(af, bf, acc[t]);
    }
  }

#pragma unroll
  for (int t = 0; t < 8; ++t) {
    const float b = bias[colBase + t * 16 + ln16];
#pragma unroll
    for (int r = 0; r < 8; ++r) acc[t][r] += b;
  }

  const int rowBase = rowBlk + wave * 16;

  if constexpr (EPI <= 2) {
#pragma unroll
    for (int t = 0; t < 8; ++t) {
      const int col = colBase + t * 16 + ln16;
#pragma unroll
      for (int r = 0; r < 8; ++r) {
        const int row = rowBase + r + 8 * hb;
        float y = acc[t][r];
        if constexpr (EPI == 1) y = fmaxf(y, 0.0f);
        if constexpr (EPI == 2)
          y = 0.5f * y * (1.0f + erff(y * 0.70710678118654752f));
        out16[(size_t)row * ldOut + col] = (_Float16)y;
      }
    }
  } else {  // row-wise LayerNorm epilogues (Nout == 128, colBase == 0)
#pragma unroll
    for (int r = 0; r < 8; ++r) {
      const int row = rowBase + r + 8 * hb;
      if constexpr (EPI == 4) {
#pragma unroll
        for (int t = 0; t < 8; ++t)
          acc[t][r] += resid[(size_t)row * DDIM + t * 16 + ln16];
      }
      float s = 0.f;
#pragma unroll
      for (int t = 0; t < 8; ++t) s += acc[t][r];
      s = red16_sum(s);
      const float mean = s * (1.0f / 128.0f);
      float vs = 0.f;
#pragma unroll
      for (int t = 0; t < 8; ++t) {
        const float d = acc[t][r] - mean;
        vs += d * d;
      }
      vs = red16_sum(vs);
      const float rstd = rsqrtf(vs * (1.0f / 128.0f) + LNEPS);
#pragma unroll
      for (int t = 0; t < 8; ++t) {
        const int col = t * 16 + ln16;
        float yn = (acc[t][r] - mean) * rstd * lng[col] + lnb[col];
        if constexpr (EPI == 3) {
          yn = (yn >= 0.f) ? yn : NSLOPE * yn;
          out32[(size_t)row * DDIM + col] = resid[(size_t)row * DDIM + col] + yn;
        } else {
          out32[(size_t)row * DDIM + col] = yn;
          if (out16) out16[(size_t)row * DDIM + col] = (_Float16)yn;
        }
      }
    }
  }
}

// ---------------------------------------------------------------- flash attention
// grid = B*NH*4 blocks; 8 waves/block; wave owns 16 queries, online softmax over
// 32-key chunks; S = Q K^T and O += P V both on v_wmma_f32_16x16x32_f16.
// K chunk staged by the TDM; V chunk transposed while staging (manual).
__global__ __launch_bounds__(256) void k_attn(const _Float16* __restrict__ qkv,
                                              _Float16* __restrict__ outp) {
  __shared__ _Float16 Kc[32 * 40];     // [key][dh]
  __shared__ _Float16 Vt[32 * 40];     // [dh][key] (pre-transposed)
  __shared__ _Float16 Ps[8][16 * 32];  // per-wave P transpose scratch
  const int lane = threadIdx.x & 31, wave = threadIdx.x >> 5;
  const int hb = lane >> 4, ln16 = lane & 15;
  const int qblk = blockIdx.x & 3;
  const int bh = blockIdx.x >> 2;
  const int b = bh >> 2, hh = bh & 3;  // NH = 4
  const int nodeBase = b * NPGR;
  const int qStart = qblk * 128 + wave * 16;

  // Q fragment straight from global (A layout; dword-aligned)
  union { v16h v; unsigned u[8]; } qa;
  {
    const _Float16* qp =
        qkv + (size_t)(nodeBase + qStart + ln16) * 384 + hh * 32;
    const int kb = hb * 8;
    const unsigned* a0 = (const unsigned*)(qp + kb);
    const unsigned* a1 = (const unsigned*)(qp + kb + 16);
#pragma unroll
    for (int i = 0; i < 4; ++i) { qa.u[i] = a0[i]; qa.u[4 + i] = a1[i]; }
  }

  float mrow[8], lrow[8];
#pragma unroll
  for (int r = 0; r < 8; ++r) { mrow[r] = -1e30f; lrow[r] = 0.f; }
  v8f O0 = {}, O1 = {};
  const float scale = 0.17677669529663687f;  // 1/sqrt(32)

  for (int kc = 0; kc < NPGR; kc += 32) {
#if USE_TDM
    if (wave == 0)
      tdm_load_2d(lds_off(Kc),
                  qkv + (size_t)(nodeBase + kc) * 384 + 128 + hh * 32, 32, 384);
#endif
    {  // stage transposed V chunk (and K when no TDM)
      const int idx = threadIdx.x * 4;
      const int key = idx >> 5, d0 = idx & 31;
      const _Float16* vp =
          qkv + (size_t)(nodeBase + kc + key) * 384 + 256 + hh * 32 + d0;
#pragma unroll
      for (int i = 0; i < 4; ++i) Vt[(d0 + i) * 40 + key] = vp[i];
#if !USE_TDM
      const _Float16* kp =
          qkv + (size_t)(nodeBase + kc + key) * 384 + 128 + hh * 32 + d0;
#pragma unroll
      for (int i = 0; i < 4; ++i) Kc[key * 40 + d0 + i] = kp[i];
#endif
    }
#if USE_TDM
    if (wave == 0) __builtin_amdgcn_s_wait_tensorcnt(0);
#endif
    __syncthreads();

    v8f S0 = {}, S1 = {};
    S0 = wmma16(qa.v, ld_bfrag(Kc, 40, lane, 0), S0);
    S1 = wmma16(qa.v, ld_bfrag(Kc, 40, lane, 16), S1);

    _Float16* ps = &Ps[wave][0];
#pragma unroll
    for (int r = 0; r < 8; ++r) {
      const float s0 = S0[r] * scale, s1 = S1[r] * scale;
      const float mx = red16_max(fmaxf(s0, s1));
      const float mnew = fmaxf(mrow[r], mx);
      const float p0 = __expf(s0 - mnew), p1 = __expf(s1 - mnew);
      const float corr = __expf(mrow[r] - mnew);
      lrow[r] = lrow[r] * corr + red16_sum(p0 + p1);
      mrow[r] = mnew;
      O0[r] *= corr;
      O1[r] *= corr;
      ps[(r + 8 * hb) * 32 + ln16] = (_Float16)p0;
      ps[(r + 8 * hb) * 32 + 16 + ln16] = (_Float16)p1;
    }
    // P (now in A layout via LDS) x V chunk
    const v16h pa = ld_afrag(ps, 32, lane);
    O0 = wmma16(pa, ld_bfrag(Vt, 40, lane, 0), O0);
    O1 = wmma16(pa, ld_bfrag(Vt, 40, lane, 16), O1);
    __syncthreads();
  }

#pragma unroll
  for (int r = 0; r < 8; ++r) {
    const int node = nodeBase + qStart + r + 8 * hb;
    const float inv = 1.0f / lrow[r];
    outp[(size_t)node * DDIM + hh * 32 + ln16] = (_Float16)(O0[r] * inv);
    outp[(size_t)node * DDIM + hh * 32 + 16 + ln16] = (_Float16)(O1[r] * inv);
  }
}

// ---------------------------------------------------------------- host
extern "C" void kernel_launch(void* const* d_in, const int* in_sizes, int n_in,
                              void* d_out, int out_size, void* d_ws,
                              size_t ws_size, hipStream_t stream) {
  (void)in_sizes; (void)n_in; (void)out_size; (void)ws_size;
  const float* x         = (const float*)d_in[0];
  const float* edge_attr = (const float*)d_in[1];
  const float* node_W    = (const float*)d_in[2];
  const float* node_b    = (const float*)d_in[3];
  const float* ep_W1     = (const float*)d_in[4];
  const float* ep_b1     = (const float*)d_in[5];
  const float* ep_W2     = (const float*)d_in[6];
  const float* ep_b2     = (const float*)d_in[7];
  const float* conv_W1   = (const float*)d_in[8];
  const float* conv_b1   = (const float*)d_in[9];
  const float* conv_W2   = (const float*)d_in[10];
  const float* conv_b2   = (const float*)d_in[11];
  const float* norm_g    = (const float*)d_in[12];
  const float* norm_b    = (const float*)d_in[13];
  const float* qkv_W     = (const float*)d_in[14];
  const float* qkv_b     = (const float*)d_in[15];
  const float* out_W     = (const float*)d_in[16];
  const float* out_b     = (const float*)d_in[17];
  const float* ln1_g     = (const float*)d_in[18];
  const float* ln1_b     = (const float*)d_in[19];
  const float* ffn_W1    = (const float*)d_in[20];
  const float* ffn_b1    = (const float*)d_in[21];
  const float* ffn_W2    = (const float*)d_in[22];
  const float* ffn_b2    = (const float*)d_in[23];
  const float* ln2_g     = (const float*)d_in[24];
  const float* ln2_b     = (const float*)d_in[25];
  const int* edge_index  = (const int*)d_in[26];
  float* outp            = (float*)d_out;

  char* base = (char*)d_ws;
  size_t off = 0;
  auto alloc = [&](size_t bytes) -> void* {
    void* p = base + off;
    off = (off + bytes + 255) & ~(size_t)255;
    return p;
  };
  _Float16* wEpW2   = (_Float16*)alloc((size_t)16384 * 2);
  _Float16* wConvW1 = (_Float16*)alloc((size_t)65536 * 2);
  _Float16* wConvW2 = (_Float16*)alloc((size_t)65536 * 2);
  _Float16* wQkv    = (_Float16*)alloc((size_t)49152 * 2);
  _Float16* wOut    = (_Float16*)alloc((size_t)16384 * 2);
  _Float16* wFfn1   = (_Float16*)alloc((size_t)65536 * 2);
  _Float16* wFfn2   = (_Float16*)alloc((size_t)65536 * 2);
  float*    hbuf    = (float*)alloc((size_t)NND * DDIM * 4);
  _Float16* e16     = (_Float16*)alloc((size_t)NEDGE * DDIM * 2);
  float*    aggr    = (float*)alloc((size_t)NND * DDIM * 4);
  _Float16* z1      = (_Float16*)alloc((size_t)NND * DDIM * 2);
  _Float16* qkv16   = (_Float16*)alloc((size_t)NND * 384 * 2);
  _Float16* attno   = (_Float16*)alloc((size_t)NND * DDIM * 2);
  float*    hp1     = (float*)alloc((size_t)NND * DDIM * 4);
  _Float16* hp1h    = (_Float16*)alloc((size_t)NND * DDIM * 2);
  _Float16* ffnh    = (_Float16*)alloc((size_t)NND * 512 * 2);

  auto cvt = [&](const float* s, _Float16* d, int n) {
    k_cvt<<<(n + 255) / 256, 256, 0, stream>>>(s, d, n);
  };
  cvt(ep_W2, wEpW2, 16384);
  cvt(conv_W1, wConvW1, 65536);
  cvt(conv_W2, wConvW2, 65536);
  cvt(qkv_W, wQkv, 49152);
  cvt(out_W, wOut, 16384);
  cvt(ffn_W1, wFfn1, 65536);
  cvt(ffn_W2, wFfn2, 65536);

  // node encoder
  k_node<<<NND / 2, 256, 0, stream>>>(x, node_W, node_b, hbuf);

  // edge encoder: fused relu(ea@W1^T+b1)@W2^T+b2 -> e16
  k_gemm<2, 0><<<dim3(NEDGE / 128, 1), 256, 0, stream>>>(
      nullptr, nullptr, nullptr, edge_attr, ep_W1, ep_b1, wEpW2, ep_b2, DDIM,
      DDIM, e16, nullptr, nullptr, nullptr, nullptr);

  // GINE conv layers
  for (int l = 0; l < NLAY; ++l) {
    k_zero4<<<(NND * DDIM / 4 + 255) / 256, 256, 0, stream>>>((float4*)aggr,
                                                              NND * DDIM / 4);
    k_msg<<<NEDGE / 8, 256, 0, stream>>>(hbuf, e16, edge_index, aggr);
    // z1 = relu((h+aggr)@W1^T+b1) -> f16
    k_gemm<1, 1><<<dim3(NND / 128, 1), 256, 0, stream>>>(
        nullptr, hbuf, aggr, nullptr, nullptr, nullptr, wConvW1 + l * 16384,
        conv_b1 + l * 128, DDIM, DDIM, z1, nullptr, nullptr, nullptr, nullptr);
    // h += leaky(LN(z1@W2^T+b2))
    k_gemm<0, 3><<<dim3(NND / 128, 1), 256, 0, stream>>>(
        z1, nullptr, nullptr, nullptr, nullptr, nullptr, wConvW2 + l * 16384,
        conv_b2 + l * 128, DDIM, DDIM, nullptr, hbuf, hbuf, norm_g + l * 128,
        norm_b + l * 128);
  }

  // QKV projection -> f16 [N,384]
  k_gemm<1, 0><<<dim3(NND / 128, 3), 256, 0, stream>>>(
      nullptr, hbuf, nullptr, nullptr, nullptr, nullptr, wQkv, qkv_b, DDIM,
      384, qkv16, nullptr, nullptr, nullptr, nullptr);

  // attention
  k_attn<<<NBAT * NHEAD * 4, 256, 0, stream>>>(qkv16, attno);

  // out projection + residual + LN1 -> hp1 (f32) and hp1h (f16)
  k_gemm<0, 4><<<dim3(NND / 128, 1), 256, 0, stream>>>(
      attno, nullptr, nullptr, nullptr, nullptr, nullptr, wOut, out_b, DDIM,
      DDIM, hp1h, hp1, hbuf, ln1_g, ln1_b);

  // FFN1: gelu -> f16 [N,512]
  k_gemm<0, 2><<<dim3(NND / 128, 4), 256, 0, stream>>>(
      hp1h, nullptr, nullptr, nullptr, nullptr, nullptr, wFfn1, ffn_b1, DDIM,
      512, ffnh, nullptr, nullptr, nullptr, nullptr);

  // FFN2 (K=512) + residual + LN2 -> final output (f32)
  k_gemm<0, 4><<<dim3(NND / 128, 1), 256, 0, stream>>>(
      ffnh, nullptr, nullptr, nullptr, nullptr, nullptr, wFfn2, ffn_b2, 512,
      DDIM, nullptr, outp, hp1, ln2_g, ln2_b);
}
